// SiamesePatchLocalizer_74242804679190
// MI455X (gfx1250) — matrix operand order repaired and checked by reading, not compiled, not run on hardware
//
#include <hip/hip_runtime.h>
#include <hip/hip_bf16.h>

typedef _Float16 f16;
typedef _Float16 v16h __attribute__((ext_vector_type(16)));
typedef _Float16 v8h  __attribute__((ext_vector_type(8)));
typedef float    v8f  __attribute__((ext_vector_type(8)));
typedef int      v4i  __attribute__((ext_vector_type(4)));

#define BN_EPS 1e-5f

// ---------------- workspace layout (bytes) ----------------
static const size_t OFF_W1F    = 0;       // 144 f32 (folded conv1 weights)
static const size_t OFF_B1F    = 1024;    // 16 f32
static const size_t OFF_W2FRAG = 2048;    // 2*5*32 v16h  (10240 B)
static const size_t OFF_B2F    = 16384;   // 32 f32
static const size_t OFF_W3FRAG = 20480;   // 4*9*32 v16h  (36864 B)
static const size_t OFF_B3F    = 61440;   // 64 f32
static const size_t OFF_ACT    = 65536;
// region A: F1S+F1P during conv1/conv2, reused as T for xcorr output
static const size_t OFF_F1S    = OFF_ACT;                             // 256*64*64*16 f16 = 32MB
static const size_t OFF_F1P    = OFF_F1S + (size_t)256*4096*16*2;     // 2MB
static const size_t OFF_T      = OFF_ACT;                             // 256*64*49*16 f32 ~ 49MB
static const size_t OFF_REGA_END = OFF_ACT + (size_t)256*64*49*16*4;
static const size_t OFF_F2S    = OFF_REGA_END;                        // 64MB
static const size_t OFF_F2P    = OFF_F2S + (size_t)256*4096*32*2;     // 4MB
static const size_t OFF_F3S    = OFF_F2P + (size_t)256*256*32*2;      // 128MB
static const size_t OFF_F3P    = OFF_F3S + (size_t)256*4096*64*2;     // 8MB
static const size_t OFF_BXF    = OFF_F3P + (size_t)256*256*64*2;      // 8MB (per-sample B fragments)

// ---------------- small helpers ----------------
__device__ __forceinline__ v16h combine16(v8h lo, v8h hi) {
  v16h a;
#pragma unroll
  for (int i = 0; i < 8; i++) { a[i] = lo[i]; a[i + 8] = hi[i]; }
  return a;
}

__device__ __forceinline__ v8h zero8() {
  v8h r;
#pragma unroll
  for (int i = 0; i < 8; i++) r[i] = (f16)0.0f;
  return r;
}

// 16-byte global -> LDS copy. Prefer CDNA5 async-to-LDS (ASYNCcnt-tracked,
// no VGPR round trip); fall back to vector load+ds_store if unavailable.
__device__ __forceinline__ void async_copy16(f16* dst_lds, const f16* src_g) {
#if __has_builtin(__builtin_amdgcn_global_load_async_to_lds_b128)
  __builtin_amdgcn_global_load_async_to_lds_b128(
      (__attribute__((address_space(1))) v4i*)(void*)src_g,
      (__attribute__((address_space(3))) v4i*)(void*)dst_lds,
      0, 0);
#else
  *(v8h*)dst_lds = *(const v8h*)src_g;
#endif
}

__device__ __forceinline__ void wait_async0() {
#if __has_builtin(__builtin_amdgcn_s_wait_asynccnt)
  __builtin_amdgcn_s_wait_asynccnt(0);
#else
  asm volatile("s_wait_asynccnt 0x0" ::: "memory");
#endif
}

// ============ prep: fold BN into conv1 weights/bias ============
__global__ void prep_w1_kernel(const float* __restrict__ w1, const float* __restrict__ b1,
                               const float* __restrict__ g1, const float* __restrict__ be1,
                               const float* __restrict__ m1, const float* __restrict__ v1,
                               float* __restrict__ w1f, float* __restrict__ b1f) {
  int c = threadIdx.x;
  if (c >= 16) return;
  float scale = g1[c] * rsqrtf(v1[c] + BN_EPS);
#pragma unroll
  for (int t = 0; t < 9; t++) w1f[c * 9 + t] = w1[c * 9 + t] * scale;
  b1f[c] = b1[c] * scale + be1[c] - m1[c] * scale;
}

// ============ prep: folded bias for layers 2/3 ============
__global__ void fold_bias_kernel(const float* __restrict__ b, const float* __restrict__ g,
                                 const float* __restrict__ be, const float* __restrict__ m,
                                 const float* __restrict__ v, float* __restrict__ out, int CO) {
  int c = blockIdx.x * blockDim.x + threadIdx.x;
  if (c >= CO) return;
  float scale = g[c] * rsqrtf(v[c] + BN_EPS);
  out[c] = b[c] * scale + be[c] - m[c] * scale;
}

// ============ prep: swizzle folded conv weights into WMMA B-fragment layout ============
// B frag packed here, contiguous k: lane<16 -> N=lane, elems = K 0..15 ; lane>=16 -> N=lane-16, elems = K 16..31
__global__ void prep_wfrag_kernel(const float* __restrict__ w, const float* __restrict__ g,
                                  const float* __restrict__ v, f16* __restrict__ dst,
                                  int CI, int Kreal, int ktiles) {
  int kt = blockIdx.x % ktiles;
  int lane = threadIdx.x;
  int co = (blockIdx.x / ktiles) * 16 + (lane & 15);
  float scale = g[co] * rsqrtf(v[co] + BN_EPS);
  int kb = kt * 32 + ((lane < 16) ? 0 : 16);
  v16h frag;
#pragma unroll
  for (int i = 0; i < 16; i++) {
    int kk = kb + i;
    float val = 0.0f;
    if (kk < Kreal) {
      int tap = kk / CI;
      int ci = kk - tap * CI;
      val = w[((size_t)co * CI + ci) * 9 + tap] * scale;
    }
    frag[i] = (f16)val;
  }
  ((v16h*)dst)[(size_t)blockIdx.x * 32 + lane] = frag;
}

// ============ conv1 (1->16) direct, BN+ReLU fused, NHWC f16 out ============
__global__ void conv1_kernel(const float* __restrict__ in, int H, int W,
                             const float* __restrict__ w1f, const float* __restrict__ b1f,
                             f16* __restrict__ out, int total) {
  int idx = blockIdx.x * blockDim.x + threadIdx.x;
  if (idx >= total) return;
  int x = idx % W;
  int y = (idx / W) % H;
  int n = idx / (W * H);
  float win[9];
#pragma unroll
  for (int dy = 0; dy < 3; dy++)
#pragma unroll
    for (int dx = 0; dx < 3; dx++) {
      int yy = y - 1 + dy, xx = x - 1 + dx;
      win[dy * 3 + dx] = (yy >= 0 && yy < H && xx >= 0 && xx < W)
                             ? in[((size_t)n * H + yy) * W + xx] : 0.0f;
    }
#pragma unroll
  for (int c = 0; c < 16; c++) {
    float acc = b1f[c];
#pragma unroll
    for (int t = 0; t < 9; t++) acc += win[t] * w1f[c * 9 + t];
    out[(size_t)idx * 16 + c] = (f16)fmaxf(acc, 0.0f);
  }
}

// ============ WMMA conv (im2col GEMM per (n,row)), BN+ReLU fused ============
// A frag as loaded here (contiguous k, same as B): lane<16 -> M=lane, K 0..15; lane>=16 -> M=lane-16, K 16..31
template <int H, int W, int CI, int CO, int Kreal, int KT>
__global__ __launch_bounds__(256) void conv_wmma_kernel(
    const f16* __restrict__ in, const v16h* __restrict__ wfrag,
    const float* __restrict__ bias, f16* __restrict__ out) {
  constexpr int ROWLEN = (W + 2) * CI;
  constexpr int CI3 = 3 * CI;
  __shared__ f16 smem[3 * ROWLEN];

  int n = blockIdx.x / H;
  int y = blockIdx.x % H;

  // stage 3 zero-padded NHWC input rows into LDS (async bulk copy for interior)
#pragma unroll
  for (int dy = 0; dy < 3; dy++) {
    int yy = y - 1 + dy;
    f16* dstrow = smem + dy * ROWLEN;
    if (yy >= 0 && yy < H) {
      const f16* srcrow = in + ((size_t)(n * H + yy) * W) * CI;
      constexpr int CHUNKS = W * CI / 8;  // 16-byte chunks
      for (int c = threadIdx.x; c < CHUNKS; c += 256)
        async_copy16(dstrow + CI + c * 8, srcrow + c * 8);
      for (int c = threadIdx.x; c < CI; c += 256) {
        dstrow[c] = (f16)0.0f;
        dstrow[(W + 1) * CI + c] = (f16)0.0f;
      }
    } else {
      for (int c = threadIdx.x; c < ROWLEN; c += 256) dstrow[c] = (f16)0.0f;
    }
  }
  wait_async0();
  __syncthreads();

  int wave = threadIdx.x >> 5;
  int lane = threadIdx.x & 31;
  constexpr int MT = W >> 4;
  constexpr int NT = CO >> 4;
  int kbase = (lane < 16) ? 0 : 16;
  int mbase = (lane < 16) ? 0 : 8;

  for (int t = wave; t < MT * NT; t += 8) {
    int mtile = t % MT;
    int nt = t / MT;
    int x = mtile * 16 + (lane & 15);
    v8f acc = {};
#pragma unroll
    for (int kt = 0; kt < KT; kt++) {
      int k0 = kt * 32 + kbase;
      int k1 = k0 + 8;
      v8h r0, r1;
      if (k0 >= Kreal) r0 = zero8();
      else { int dy = k0 / CI3; int rem = k0 - dy * CI3;
             r0 = *(const v8h*)(smem + dy * ROWLEN + x * CI + rem); }
      if (k1 >= Kreal) r1 = zero8();
      else { int dy = k1 / CI3; int rem = k1 - dy * CI3;
             r1 = *(const v8h*)(smem + dy * ROWLEN + x * CI + rem); }
      v16h a = combine16(r0, r1);
      v16h b = wfrag[((size_t)nt * KT + kt) * 32 + lane];
      acc = __builtin_amdgcn_wmma_f32_16x16x32_f16(false, a, false, b, (short)0, acc,
                                                   false, false);
    }
    int co = nt * 16 + (lane & 15);
    float bb = bias[co];
#pragma unroll
    for (int r = 0; r < 8; r++) {
      float val = fmaxf(acc[r] + bb, 0.0f);
      int xm = mtile * 16 + mbase + r;
      out[((size_t)(n * H + y) * W + xm) * CO + co] = (f16)val;
    }
  }
}

// ============ swizzle patch features into per-sample WMMA B-fragments ============
// B[k][dy] = Fp[n, dy, dx, ci], k = dx*64+ci (K=1024 -> 32 ktiles); dst[n][kt][lane]
__global__ void bxprep_kernel(const f16* __restrict__ fp, f16* __restrict__ bx) {
  int n = blockIdx.x >> 5;
  int kt = blockIdx.x & 31;
  int lane = threadIdx.x;
  int dyN = lane & 15;
  int k0 = kt * 32 + ((lane < 16) ? 0 : 16);
  int dx = k0 >> 6;
  int ci = k0 & 63;
  v16h b = *(const v16h*)(fp + (((size_t)n * 16 + dyN) * 16 + dx) * 64 + ci);
  ((v16h*)bx)[(size_t)blockIdx.x * 32 + lane] = b;
}

// ============ xcorr partial: T[n][y][x0][dy] = sum_{dx,ci} Fs[n,y,x0+dx,ci]*Fp[n,dy,dx,ci] ==========
__global__ __launch_bounds__(128) void xcorr_kernel(const f16* __restrict__ fs,
                                                    const v16h* __restrict__ bx,
                                                    float* __restrict__ T) {
  __shared__ f16 row[80 * 64];  // one source row (NHWC), zero-padded to 80 x-positions
  int n = blockIdx.x >> 6;
  int y = blockIdx.x & 63;
  const f16* src = fs + (((size_t)n * 64 + y) * 64) * 64;
  for (int c = threadIdx.x; c < 512; c += 128)          // 64*64 f16 = 512 x 16B
    async_copy16(row + c * 8, src + c * 8);
  for (int i = threadIdx.x; i < 16 * 64; i += 128)      // zero pad x = 64..79
    row[64 * 64 + i] = (f16)0.0f;
  wait_async0();
  __syncthreads();

  int mtile = threadIdx.x >> 5;  // 4 waves, one M-tile each
  int lane = threadIdx.x & 31;
  int x0 = mtile * 16 + (lane & 15);
  int kbase = (lane < 16) ? 0 : 16;

  v8f acc = {};
#pragma unroll
  for (int kt = 0; kt < 32; kt++) {
    int k0 = kt * 32 + kbase;
    v8h r0 = *(const v8h*)(row + x0 * 64 + k0);
    v8h r1 = *(const v8h*)(row + x0 * 64 + k0 + 8);
    v16h a = combine16(r0, r1);
    v16h b = bx[((size_t)n * 32 + kt) * 32 + lane];
    acc = __builtin_amdgcn_wmma_f32_16x16x32_f16(false, a, false, b, (short)0, acc,
                                                 false, false);
  }
  int dy = lane & 15;
  int mbase = (lane < 16) ? 0 : 8;
#pragma unroll
  for (int r = 0; r < 8; r++) {
    int xm = mtile * 16 + mbase + r;
    if (xm < 49)
      T[(((size_t)n * 64 + y) * 49 + xm) * 16 + dy] = acc[r];
  }
}

// ============ corr = sum_dy T[y0+dy] ; stable soft-argmax ============
__global__ __launch_bounds__(256) void reduce_softargmax_kernel(const float* __restrict__ T,
                                                                float* __restrict__ out) {
  int n = blockIdx.x;
  int tid = threadIdx.x;
  float* corr = out + 512 + (size_t)n * 2401;
  __shared__ float red[256];

  float vals[10];
  float lmax = -3.4e38f;
  int cnt = 0;
  for (int p = tid; p < 2401; p += 256) {
    int y0 = p / 49, x0 = p - y0 * 49;
    float s = 0.0f;
#pragma unroll
    for (int dy = 0; dy < 16; dy++)
      s += T[(((size_t)n * 64 + (y0 + dy)) * 49 + x0) * 16 + dy];
    corr[p] = s;
    vals[cnt++] = s;
    lmax = fmaxf(lmax, s);
  }
  red[tid] = lmax;
  __syncthreads();
  for (int off = 128; off > 0; off >>= 1) {
    if (tid < off) red[tid] = fmaxf(red[tid], red[tid + off]);
    __syncthreads();
  }
  float smax = red[0];
  __syncthreads();

  float se = 0.0f, sy = 0.0f, sx = 0.0f;
  cnt = 0;
  for (int p = tid; p < 2401; p += 256) {
    int y0 = p / 49, x0 = p - y0 * 49;
    float e = __expf((vals[cnt++] - smax) * 10.0f);
    se += e; sy += e * (float)y0; sx += e * (float)x0;
  }
  red[tid] = se; __syncthreads();
  for (int off = 128; off > 0; off >>= 1) { if (tid < off) red[tid] += red[tid + off]; __syncthreads(); }
  float tot = red[0]; __syncthreads();
  red[tid] = sy; __syncthreads();
  for (int off = 128; off > 0; off >>= 1) { if (tid < off) red[tid] += red[tid + off]; __syncthreads(); }
  float toty = red[0]; __syncthreads();
  red[tid] = sx; __syncthreads();
  for (int off = 128; off > 0; off >>= 1) { if (tid < off) red[tid] += red[tid + off]; __syncthreads(); }
  float totx = red[0];
  if (tid == 0) {
    out[n * 2 + 0] = toty / tot;
    out[n * 2 + 1] = totx / tot;
  }
}

// ---------------- launch ----------------
extern "C" void kernel_launch(void* const* d_in, const int* in_sizes, int n_in,
                              void* d_out, int out_size, void* d_ws, size_t ws_size,
                              hipStream_t stream) {
  const float* src   = (const float*)d_in[0];
  const float* patch = (const float*)d_in[1];
  const float* w1 = (const float*)d_in[2];  const float* b1 = (const float*)d_in[3];
  const float* g1 = (const float*)d_in[4];  const float* be1 = (const float*)d_in[5];
  const float* m1 = (const float*)d_in[6];  const float* v1 = (const float*)d_in[7];
  const float* w2 = (const float*)d_in[8];  const float* b2 = (const float*)d_in[9];
  const float* g2 = (const float*)d_in[10]; const float* be2 = (const float*)d_in[11];
  const float* m2 = (const float*)d_in[12]; const float* v2 = (const float*)d_in[13];
  const float* w3 = (const float*)d_in[14]; const float* b3 = (const float*)d_in[15];
  const float* g3 = (const float*)d_in[16]; const float* be3 = (const float*)d_in[17];
  const float* m3 = (const float*)d_in[18]; const float* v3 = (const float*)d_in[19];

  char* ws = (char*)d_ws;
  float* w1f = (float*)(ws + OFF_W1F);
  float* b1f = (float*)(ws + OFF_B1F);
  f16* w2frag = (f16*)(ws + OFF_W2FRAG);
  float* b2f = (float*)(ws + OFF_B2F);
  f16* w3frag = (f16*)(ws + OFF_W3FRAG);
  float* b3f = (float*)(ws + OFF_B3F);
  f16* F1S = (f16*)(ws + OFF_F1S);
  f16* F1P = (f16*)(ws + OFF_F1P);
  f16* F2S = (f16*)(ws + OFF_F2S);
  f16* F2P = (f16*)(ws + OFF_F2P);
  f16* F3S = (f16*)(ws + OFF_F3S);
  f16* F3P = (f16*)(ws + OFF_F3P);
  f16* BX  = (f16*)(ws + OFF_BXF);
  float* Tb = (float*)(ws + OFF_T);
  float* out = (float*)d_out;

  // weight prep
  prep_w1_kernel<<<1, 32, 0, stream>>>(w1, b1, g1, be1, m1, v1, w1f, b1f);
  fold_bias_kernel<<<1, 32, 0, stream>>>(b2, g2, be2, m2, v2, b2f, 32);
  fold_bias_kernel<<<1, 64, 0, stream>>>(b3, g3, be3, m3, v3, b3f, 64);
  prep_wfrag_kernel<<<2 * 5, 32, 0, stream>>>(w2, g2, v2, w2frag, 16, 144, 5);
  prep_wfrag_kernel<<<4 * 9, 32, 0, stream>>>(w3, g3, v3, w3frag, 32, 288, 9);

  // layer 1 (direct)
  conv1_kernel<<<(256 * 64 * 64) / 256, 256, 0, stream>>>(src, 64, 64, w1f, b1f, F1S, 256 * 64 * 64);
  conv1_kernel<<<(256 * 16 * 16) / 256, 256, 0, stream>>>(patch, 16, 16, w1f, b1f, F1P, 256 * 16 * 16);

  // layer 2 (WMMA): K = 9*16 = 144 -> 5 k-tiles (padded to 160)
  conv_wmma_kernel<64, 64, 16, 32, 144, 5><<<256 * 64, 256, 0, stream>>>(F1S, (const v16h*)w2frag, b2f, F2S);
  conv_wmma_kernel<16, 16, 16, 32, 144, 5><<<256 * 16, 256, 0, stream>>>(F1P, (const v16h*)w2frag, b2f, F2P);

  // layer 3 (WMMA): K = 9*32 = 288 -> 9 k-tiles (exact)
  conv_wmma_kernel<64, 64, 32, 64, 288, 9><<<256 * 64, 256, 0, stream>>>(F2S, (const v16h*)w3frag, b3f, F3S);
  conv_wmma_kernel<16, 16, 32, 64, 288, 9><<<256 * 16, 256, 0, stream>>>(F2P, (const v16h*)w3frag, b3f, F3P);

  // per-sample xcorr (WMMA) + soft-argmax
  bxprep_kernel<<<256 * 32, 32, 0, stream>>>(F3P, BX);
  xcorr_kernel<<<256 * 64, 128, 0, stream>>>(F3S, (const v16h*)BX, Tb);
  reduce_softargmax_kernel<<<256, 256, 0, stream>>>(Tb, out);
}
